// RDE_71365176590638
// MI455X (gfx1250) — matrix operand order repaired and checked by reading, not compiled, not run on hardware
//
#include <hip/hip_runtime.h>
#include <hip/hip_bf16.h>

typedef __attribute__((ext_vector_type(16))) _Float16 v16h;
typedef __attribute__((ext_vector_type(8)))  float    v8f;

#define M_TOTAL 8192
#define N_STEPS 256
#define H_DIM   256
#define SIG_DIM 62
#define K_PAD   64
#define MBLK    64
#define NTILES  16
#define THREADS 128

// ---- wave-level MLP: h = relu(zz @ W1 + b1); u = h @ W2 + b2 -----------------
// ZZ:   (MBLK x K_PAD) f16, row-major in LDS (A matrices)
// W1h:  pre-swizzled B fragments [ntile][kchunk][lane][16 halves]
__device__ __forceinline__ void mlp_wave(const _Float16* ZZ, const _Float16* W1h,
                                         const float* B1s, const float* W2s,
                                         float b2v, float* u_sh, int tid)
{
    const int w  = tid >> 5;      // wave id 0..3 -> 16-row tile
    const int L  = tid & 31;      // lane
    const int hs = L >> 4;        // half-select (A: K-half / C: M-half)
    const int nl = L & 15;        // N-lane / A-row

    // --- A fragments: 16x32 f16 layout: lanes0-15 K{0..7,16..23}, lanes16-31 K{8..15,24..31}
    v16h a0, a1;
    {
        const _Float16* zr = ZZ + (w * 16 + nl) * K_PAD;
        const int k0 = hs * 8;
        float4 s0 = *(const float4*)(zr + k0);
        float4 s1 = *(const float4*)(zr + 16 + k0);
        float4 s2 = *(const float4*)(zr + 32 + k0);
        float4 s3 = *(const float4*)(zr + 48 + k0);
        __builtin_memcpy(&a0, &s0, 16); __builtin_memcpy((char*)&a0 + 16, &s1, 16);
        __builtin_memcpy(&a1, &s2, 16); __builtin_memcpy((char*)&a1 + 16, &s3, 16);
    }

    float up[8] = {0.f, 0.f, 0.f, 0.f, 0.f, 0.f, 0.f, 0.f};

    #pragma unroll
    for (int nt = 0; nt < NTILES; ++nt) {
        // --- B fragments: 32x16 f16, lane holds column N=L%16, K-half = L/16
        v16h bA, bB;
        const _Float16* bp0 = W1h + ((nt * 2 + 0) * 32 + L) * 16;
        const _Float16* bp1 = W1h + ((nt * 2 + 1) * 32 + L) * 16;
        float4 t0 = *(const float4*)(bp0);
        float4 t1 = *(const float4*)(bp0 + 8);
        float4 t2 = *(const float4*)(bp1);
        float4 t3 = *(const float4*)(bp1 + 8);
        __builtin_memcpy(&bA, &t0, 16); __builtin_memcpy((char*)&bA + 16, &t1, 16);
        __builtin_memcpy(&bB, &t2, 16); __builtin_memcpy((char*)&bB + 16, &t3, 16);

        v8f c = {};
        c = __builtin_amdgcn_wmma_f32_16x16x32_f16(false, a0, false, bA, (short)0, c, false, false);
        c = __builtin_amdgcn_wmma_f32_16x16x32_f16(false, a1, false, bB, (short)0, c, false, false);

        const float bb = B1s[nt * 16 + nl];
        const float ww = W2s[nt * 16 + nl];
        #pragma unroll
        for (int v = 0; v < 8; ++v) {
            float h = c[v] + bb;           // C layout: row = w*16 + hs*8 + v, col = nt*16 + nl
            h = h > 0.f ? h : 0.f;         // relu
            up[v] = fmaf(h, ww, up[v]);    // dot with W2 column segment
        }
    }

    // reduce over the 16 N-lanes within each half-wave group (wave32 xor shuffles)
    #pragma unroll
    for (int m = 1; m < 16; m <<= 1) {
        #pragma unroll
        for (int v = 0; v < 8; ++v) up[v] += __shfl_xor(up[v], m, 32);
    }
    if (nl == 0) {
        const int rb = w * 16 + hs * 8;
        #pragma unroll
        for (int v = 0; v < 8; ++v) u_sh[rb + v] = up[v] + b2v;
    }
}

__global__ __launch_bounds__(THREADS)
void sigrde_scan_kernel(const float* __restrict__ X,
                        const float* __restrict__ W1,
                        const float* __restrict__ B1,
                        const float* __restrict__ W2,
                        const float* __restrict__ B2,
                        float* __restrict__ Y,
                        float* __restrict__ U)
{
    __shared__ alignas(16) _Float16 W1h[NTILES * 2 * 32 * 16]; // 32 KB: B fragments
    __shared__ alignas(16) _Float16 ZZ[MBLK * K_PAD];          // 8 KB: A matrices
    __shared__ float W2s[H_DIM];
    __shared__ float B1s[H_DIM];
    __shared__ float u_sh[MBLK];

    const int tid = threadIdx.x;
    const int row = blockIdx.x * MBLK + tid;   // valid only for owner threads
    const bool owner = (tid < MBLK);

    // ---- one-time: swizzle W1 (62x256 f32 row-major) into B-fragment LDS layout
    for (int t = tid; t < NTILES * 2 * 32; t += THREADS) {
        const int lane  = t & 31;
        const int kc    = (t >> 5) & 1;
        const int nt    = t >> 6;
        const int nn    = nt * 16 + (lane & 15);
        const int kbase = kc * 32 + (lane >> 4) * 8;
        _Float16* dst = &W1h[t * 16];
        #pragma unroll
        for (int j = 0; j < 16; ++j) {
            const int k = kbase + (j < 8 ? j : j + 8);
            const float v = (k < SIG_DIM) ? W1[k * H_DIM + nn] : 0.0f;
            dst[j] = (_Float16)v;
        }
    }
    for (int t = tid; t < H_DIM; t += THREADS) { W2s[t] = W2[t]; B1s[t] = B1[t]; }
    const float b2v = B2[0];

    // ---- per-row signature state in registers
    float L1v[2] = {0.f, 0.f};
    float L2v[4] = {0.f, 0.f, 0.f, 0.f};
    float L3v[8] = {0.f};
    float L4v[16] = {0.f};
    float L5v[32] = {0.f};
    float yv = 0.f, xp0 = 0.f, xp1 = 0.f;

    if (owner) {
        const size_t xb = (size_t)row * N_STEPS * 2;
        xp0 = X[xb + 0];
        xp1 = X[xb + 1];
        Y[(size_t)row * N_STEPS + 0] = 0.f;
        _Float16* zz = &ZZ[tid * K_PAD];
        #pragma unroll
        for (int j = 0; j < K_PAD; ++j) zz[j] = (_Float16)0.f;   // pad lanes stay 0 forever
    }

    for (int i = 0; i < N_STEPS - 1; ++i) {
        __syncthreads();                                  // ZZ ready for this step
        mlp_wave(ZZ, W1h, B1s, W2s, b2v, u_sh, tid);
        __syncthreads();                                  // u ready; A-frag reads done

        if (owner) {
            const float u = u_sh[tid];
            const size_t xb = (size_t)row * N_STEPS * 2 + (size_t)(i + 1) * 2;
            const float xc0 = X[xb + 0];
            const float xc1 = X[xb + 1];
            const float dz0 = xc0 - xp0, dz1 = xc1 - xp1;
            xp0 = xc0; xp1 = xc1;

            yv = fmaf(u, dz1, yv);
            Y[(size_t)row * N_STEPS + i + 1] = yv;
            U[(size_t)row * N_STEPS + i]     = u;

            // ---- Chen join: Sig <- Sig (x) exp(dz), truncated at level 5
            float E1v[2] = {dz0, dz1};
            float E2v[4], E3v[8], E4v[16], E5v[32];
            #pragma unroll
            for (int a = 0; a < 2; ++a)
                #pragma unroll
                for (int b = 0; b < 2; ++b) E2v[a*2+b] = E1v[a] * E1v[b] * 0.5f;
            #pragma unroll
            for (int a = 0; a < 4; ++a)
                #pragma unroll
                for (int b = 0; b < 2; ++b) E3v[a*2+b] = E2v[a] * E1v[b] * (1.0f/3.0f);
            #pragma unroll
            for (int a = 0; a < 8; ++a)
                #pragma unroll
                for (int b = 0; b < 2; ++b) E4v[a*2+b] = E3v[a] * E1v[b] * 0.25f;
            #pragma unroll
            for (int a = 0; a < 16; ++a)
                #pragma unroll
                for (int b = 0; b < 2; ++b) E5v[a*2+b] = E4v[a] * E1v[b] * 0.2f;

            // update high->low: level k reads only lower (old) levels, safe in place
            #pragma unroll
            for (int j = 0; j < 32; ++j) L5v[j] += E5v[j];
            #pragma unroll
            for (int a = 0; a < 2; ++a)
                #pragma unroll
                for (int b = 0; b < 16; ++b) L5v[a*16+b] = fmaf(L1v[a], E4v[b], L5v[a*16+b]);
            #pragma unroll
            for (int a = 0; a < 4; ++a)
                #pragma unroll
                for (int b = 0; b < 8; ++b)  L5v[a*8+b]  = fmaf(L2v[a], E3v[b], L5v[a*8+b]);
            #pragma unroll
            for (int a = 0; a < 8; ++a)
                #pragma unroll
                for (int b = 0; b < 4; ++b)  L5v[a*4+b]  = fmaf(L3v[a], E2v[b], L5v[a*4+b]);
            #pragma unroll
            for (int a = 0; a < 16; ++a)
                #pragma unroll
                for (int b = 0; b < 2; ++b)  L5v[a*2+b]  = fmaf(L4v[a], E1v[b], L5v[a*2+b]);

            #pragma unroll
            for (int j = 0; j < 16; ++j) L4v[j] += E4v[j];
            #pragma unroll
            for (int a = 0; a < 2; ++a)
                #pragma unroll
                for (int b = 0; b < 8; ++b)  L4v[a*8+b]  = fmaf(L1v[a], E3v[b], L4v[a*8+b]);
            #pragma unroll
            for (int a = 0; a < 4; ++a)
                #pragma unroll
                for (int b = 0; b < 4; ++b)  L4v[a*4+b]  = fmaf(L2v[a], E2v[b], L4v[a*4+b]);
            #pragma unroll
            for (int a = 0; a < 8; ++a)
                #pragma unroll
                for (int b = 0; b < 2; ++b)  L4v[a*2+b]  = fmaf(L3v[a], E1v[b], L4v[a*2+b]);

            #pragma unroll
            for (int j = 0; j < 8; ++j) L3v[j] += E3v[j];
            #pragma unroll
            for (int a = 0; a < 2; ++a)
                #pragma unroll
                for (int b = 0; b < 4; ++b)  L3v[a*4+b]  = fmaf(L1v[a], E2v[b], L3v[a*4+b]);
            #pragma unroll
            for (int a = 0; a < 4; ++a)
                #pragma unroll
                for (int b = 0; b < 2; ++b)  L3v[a*2+b]  = fmaf(L2v[a], E1v[b], L3v[a*2+b]);

            #pragma unroll
            for (int j = 0; j < 4; ++j) L2v[j] += E2v[j];
            #pragma unroll
            for (int a = 0; a < 2; ++a)
                #pragma unroll
                for (int b = 0; b < 2; ++b)  L2v[a*2+b]  = fmaf(L1v[a], E1v[b], L2v[a*2+b]);

            L1v[0] += dz0; L1v[1] += dz1;

            // ---- publish updated zz (f16) for next step's WMMA
            _Float16* zz = &ZZ[tid * K_PAD];
            #pragma unroll
            for (int j = 0; j < 2; ++j)  zz[j]      = (_Float16)L1v[j];
            #pragma unroll
            for (int j = 0; j < 4; ++j)  zz[2 + j]  = (_Float16)L2v[j];
            #pragma unroll
            for (int j = 0; j < 8; ++j)  zz[6 + j]  = (_Float16)L3v[j];
            #pragma unroll
            for (int j = 0; j < 16; ++j) zz[14 + j] = (_Float16)L4v[j];
            #pragma unroll
            for (int j = 0; j < 32; ++j) zz[30 + j] = (_Float16)L5v[j];
        }
    }

    // final strategy evaluation: u_last = strat(final levels)
    __syncthreads();
    mlp_wave(ZZ, W1h, B1s, W2s, b2v, u_sh, tid);
    __syncthreads();
    if (owner) U[(size_t)row * N_STEPS + (N_STEPS - 1)] = u_sh[tid];
}

extern "C" void kernel_launch(void* const* d_in, const int* in_sizes, int n_in,
                              void* d_out, int out_size, void* d_ws, size_t ws_size,
                              hipStream_t stream) {
    const float* X  = (const float*)d_in[0];
    const float* W1 = (const float*)d_in[1];
    const float* B1 = (const float*)d_in[2];
    const float* W2 = (const float*)d_in[3];
    const float* B2 = (const float*)d_in[4];
    float* Y = (float*)d_out;
    float* U = Y + (size_t)M_TOTAL * N_STEPS;

    dim3 grid(M_TOTAL / MBLK);   // 128 blocks, each owns 64 rows for all 255 steps
    dim3 block(THREADS);         // 4 waves (wave32): one 16-row WMMA tile per wave
    hipLaunchKernelGGL(sigrde_scan_kernel, grid, block, 0, stream,
                       X, W1, B1, W2, B2, Y, U);
}
